// DeepSeekMoE_11785390260703
// MI455X (gfx1250) — matrix operand order repaired and checked by reading, not compiled, not run on hardware
//
#include <hip/hip_runtime.h>
#include <hip/hip_bf16.h>
#include <math.h>

// ---------------- CDNA5 WMMA vector types ----------------
typedef __attribute__((ext_vector_type(16))) __bf16 v16bf;
typedef __attribute__((ext_vector_type(8)))  float  v8f;

// ---------------- problem constants (from reference) ----------------
#define H_DIM   2048
#define I_DIM   1024
#define T_TOK   4096      // B*S
#define NEXP    10        // 2 shared + 8 routed slots
#define NROUTED 8

// ---------------- tiling ----------------
#define TM  128           // macro tile M (tokens)
#define TN  64            // macro tile N
#define TK  32            // K chunk (bf16 wmma K)
#define LDA 48            // padded LDS K-stride (96B rows, 16B-aligned chunks)

struct U4 { unsigned int a, b, c, d; };
struct U2 { unsigned int a, b; };
struct FragPair { U4 lo, hi; };

__device__ __forceinline__ unsigned short f32_bf16(float f) {
  unsigned u = __float_as_uint(f);
  u += 0x7FFFu + ((u >> 16) & 1u);      // round-to-nearest-even
  return (unsigned short)(u >> 16);
}

__device__ __forceinline__ v16bf make_frag(const unsigned short* p0,
                                           const unsigned short* p1) {
  FragPair t;
  t.lo = *(const U4*)p0;
  t.hi = *(const U4*)p1;
  return __builtin_bit_cast(v16bf, t);
}

// CDNA5 async data mover: copy 16B global -> LDS without touching VGPR data.
// Generic shared pointers on gfx1250 carry the LDS byte offset in the low 32
// bits (aperture bits live in [63:32]), so truncation yields the DS address.
__device__ __forceinline__ void async_b128(const unsigned short* lds,
                                           const unsigned short* g) {
  unsigned l = (unsigned)(size_t)lds;
  unsigned long long a = (unsigned long long)(size_t)g;
  asm volatile("global_load_async_to_lds_b128 %0, %1, off"
               :: "v"(l), "v"(a) : "memory");
}
// Async loads retire in order per wave: waiting ASYNCcnt <= N retires the
// oldest outstanding stages while keeping N instructions in flight.
template <int N>
__device__ __forceinline__ void wait_async_le() {
  asm volatile("s_wait_asynccnt %0" :: "i"(N) : "memory");
}

// ================= kernel 1: router (sigmoid top-2) =================
__global__ __launch_bounds__(256)
void moe_router_k(const float* __restrict__ x,
                  const float* __restrict__ rw,      // [H, 8]
                  const float* __restrict__ rb,      // [8]
                  float* __restrict__ comb,          // [T, 8]
                  float* __restrict__ usage) {       // [8]
  const int wave = threadIdx.x >> 5;
  const int lane = threadIdx.x & 31;
  const int t = blockIdx.x * 8 + wave;
  if (t >= T_TOK) return;

  float acc[8] = {0.f, 0.f, 0.f, 0.f, 0.f, 0.f, 0.f, 0.f};
  const float* xr = x + (size_t)t * H_DIM;
  for (int h = lane; h < H_DIM; h += 32) {
    float xv = xr[h];
    const float4* wr = (const float4*)(rw + (size_t)h * 8);
    float4 a = wr[0], b = wr[1];
    acc[0] += xv * a.x; acc[1] += xv * a.y; acc[2] += xv * a.z; acc[3] += xv * a.w;
    acc[4] += xv * b.x; acc[5] += xv * b.y; acc[6] += xv * b.z; acc[7] += xv * b.w;
  }
#pragma unroll
  for (int off = 16; off >= 1; off >>= 1)
#pragma unroll
    for (int e = 0; e < 8; ++e)
      acc[e] += __shfl_xor(acc[e], off, 32);

  if (lane == 0) {
    float p[8];
#pragma unroll
    for (int e = 0; e < 8; ++e)
      p[e] = 1.0f / (1.0f + expf(-(acc[e] + rb[e])));
    int i1 = 0;
#pragma unroll
    for (int e = 1; e < 8; ++e) if (p[e] > p[i1]) i1 = e;   // ties -> lower idx
    int i2 = (i1 == 0) ? 1 : 0;
#pragma unroll
    for (int e = 0; e < 8; ++e) if (e != i1 && p[e] > p[i2]) i2 = e;
    float s = p[i1] + p[i2];
    float* c = comb + (size_t)t * 8;
#pragma unroll
    for (int e = 0; e < 8; ++e) c[e] = 0.0f;
    c[i1] = p[i1] / s;
    c[i2] = p[i2] / s;
    atomicAdd(usage + i1, 1.0f);
    atomicAdd(usage + i2, 1.0f);
  }
}

// ================= kernel 2: x fp32 -> bf16 =================
__global__ __launch_bounds__(256)
void conv_bf16_k(const float* __restrict__ x, unsigned short* __restrict__ xb, int n4) {
  int i = blockIdx.x * blockDim.x + threadIdx.x;
  if (i >= n4) return;
  float4 v = ((const float4*)x)[i];
  U2 o;
  o.a = (unsigned)f32_bf16(v.x) | ((unsigned)f32_bf16(v.y) << 16);
  o.b = (unsigned)f32_bf16(v.z) | ((unsigned)f32_bf16(v.w) << 16);
  ((U2*)xb)[i] = o;
}

// ===== kernel 2b: weight transpose+convert: fp32 [K][N] -> bf16 [N][K] =====
// grid: (N/32, K/32, NEXP); block 256 (32x8)
__global__ __launch_bounds__(256)
void wtrans_k(const float* __restrict__ sW,   // [2,K,N]
              const float* __restrict__ rW,   // [8,K,N]
              unsigned short* __restrict__ dst,   // [NEXP,N,K]
              int K, int N) {
  __shared__ float tile[32][33];
  const int e = blockIdx.z;
  const float* src = (e < 2) ? sW + (size_t)e * K * N
                             : rW + (size_t)(e - 2) * K * N;
  unsigned short* d = dst + (size_t)e * K * N;
  const int n0 = blockIdx.x * 32, k0 = blockIdx.y * 32;
  const int tx = threadIdx.x & 31, ty = threadIdx.x >> 5;
#pragma unroll
  for (int j = 0; j < 32; j += 8)
    tile[ty + j][tx] = src[(size_t)(k0 + ty + j) * N + n0 + tx];
  __syncthreads();
#pragma unroll
  for (int j = 0; j < 32; j += 8)
    d[(size_t)(n0 + ty + j) * K + k0 + tx] = f32_bf16(tile[tx][ty + j]);
}

// ===== kernel 3: fused gate+up GEMM + SwiGLU + routing-weight fold =====
// grid: (I/TN, T/TM, NEXP), block 256 (8 waves, each 32x32)
// 3 LDS buffers, prefetch distance 2, 4 async issues per stage.
__global__ __launch_bounds__(256)
void moe_gemm1_k(const unsigned short* __restrict__ xb,     // [T,H] bf16
                 const unsigned short* __restrict__ wtg,    // [NEXP,I,H] bf16
                 const unsigned short* __restrict__ wtu,    // [NEXP,I,H] bf16
                 const float* __restrict__ comb,            // [T,8]
                 unsigned short* __restrict__ hb) {         // [NEXP,T,I] bf16
  __shared__ __attribute__((aligned(16))) unsigned short As[3][TM][LDA];
  __shared__ __attribute__((aligned(16))) unsigned short Bg[3][TN][LDA];
  __shared__ __attribute__((aligned(16))) unsigned short Bu[3][TN][LDA];

  const int e  = blockIdx.z;
  const int n0 = blockIdx.x * TN;
  const int t0 = blockIdx.y * TM;
  const int tid  = threadIdx.x;
  const int lane = tid & 31;
  const int wm   = (tid >> 5) >> 1;   // 0..3
  const int wn   = (tid >> 5) & 1;    // 0..1

  const int ar = tid >> 1, ah = (tid & 1) * 16;   // A: 2 threads/row
  const int br = tid >> 2, bq = (tid & 3) * 8;    // B: 4 threads/row

  const unsigned short* aSrc = xb  + (size_t)(t0 + ar) * H_DIM + ah;
  const unsigned short* gSrc = wtg + (size_t)e * H_DIM * I_DIM
                                   + (size_t)(n0 + br) * H_DIM + bq;
  const unsigned short* uSrc = wtu + (size_t)e * H_DIM * I_DIM
                                   + (size_t)(n0 + br) * H_DIM + bq;

  v8f accG[2][2] = {};
  v8f accU[2][2] = {};

  const int ml = lane & 15;
  const int kh = lane >> 4;

  auto stage = [&](int buf, int c) {
    const int k = c * TK;
    async_b128(&As[buf][ar][ah],     aSrc + k);
    async_b128(&As[buf][ar][ah + 8], aSrc + k + 8);
    async_b128(&Bg[buf][br][bq],     gSrc + k);
    async_b128(&Bu[buf][br][bq],     uSrc + k);
  };
  auto compute = [&](int buf) {
    v16bf aF[2], gF[2], uF[2];
#pragma unroll
    for (int ms = 0; ms < 2; ++ms) {
      const unsigned short* p = &As[buf][wm * 32 + ms * 16 + ml][kh * 16];
      aF[ms] = make_frag(p, p + 8);
    }
#pragma unroll
    for (int ns = 0; ns < 2; ++ns) {
      const unsigned short* pg = &Bg[buf][wn * 32 + ns * 16 + ml][kh * 16];
      gF[ns] = make_frag(pg, pg + 8);
      const unsigned short* pu = &Bu[buf][wn * 32 + ns * 16 + ml][kh * 16];
      uF[ns] = make_frag(pu, pu + 8);
    }
#pragma unroll
    for (int ms = 0; ms < 2; ++ms)
#pragma unroll
      for (int ns = 0; ns < 2; ++ns) {
        accG[ms][ns] = __builtin_amdgcn_wmma_f32_16x16x32_bf16(
            false, aF[ms], false, gF[ns], (short)0, accG[ms][ns], false, false);
        accU[ms][ns] = __builtin_amdgcn_wmma_f32_16x16x32_bf16(
            false, aF[ms], false, uF[ns], (short)0, accU[ms][ns], false, false);
      }
  };

  const int NC = H_DIM / TK;   // 64
  stage(0, 0);
  stage(1, 1);
  int c = 0;
  for (; c + 2 < NC; ++c) {            // steady state: 2 stages in flight
    wait_async_le<4>();                // oldest stage (chunk c) has landed
    __syncthreads();                   // all waves' chunk c landed; old reads done
    stage((c + 2) % 3, c + 2);         // prefetch 2 ahead
    compute(c % 3);
  }
  for (; c < NC; ++c) {                // tail: full drain
    wait_async_le<0>();
    __syncthreads();
    compute(c % 3);
  }

  // epilogue: h = silu(g)*u, fold routing weight, store bf16
  const int nl = lane & 15;
  const int mb = (lane >> 4) * 8;
  unsigned short* hE = hb + (size_t)e * T_TOK * I_DIM;
#pragma unroll
  for (int ms = 0; ms < 2; ++ms)
#pragma unroll
    for (int ns = 0; ns < 2; ++ns) {
      int col = n0 + wn * 32 + ns * 16 + nl;
#pragma unroll
      for (int r = 0; r < 8; ++r) {
        int t = t0 + wm * 32 + ms * 16 + mb + r;
        float g = accG[ms][ns][r];
        float u = accU[ms][ns][r];
        float h = (g / (1.0f + expf(-g))) * u;
        float w = (e < 2) ? 1.0f : comb[(size_t)t * 8 + (e - 2)];
        hE[(size_t)t * I_DIM + col] = f32_bf16(h * w);
      }
    }
}

// ===== kernel 4: down-projection GEMM, K = NEXP*I (experts folded in K) =====
// grid: (H/TN, T/TM), block 256
// 4 LDS buffers, prefetch distance 3, 3 async issues per stage.
__global__ __launch_bounds__(256)
void moe_gemm2_k(const unsigned short* __restrict__ hb,    // [NEXP,T,I] bf16
                 const unsigned short* __restrict__ wtd,   // [NEXP,H,I] bf16
                 float* __restrict__ out) {                // [T,H]
  __shared__ __attribute__((aligned(16))) unsigned short As[4][TM][LDA];
  __shared__ __attribute__((aligned(16))) unsigned short Bs[4][TN][LDA];

  const int n0 = blockIdx.x * TN;
  const int t0 = blockIdx.y * TM;
  const int tid  = threadIdx.x;
  const int lane = tid & 31;
  const int wm   = (tid >> 5) >> 1;
  const int wn   = (tid >> 5) & 1;

  const int ar = tid >> 1, ah = (tid & 1) * 16;
  const int br = tid >> 2, bq = (tid & 3) * 8;
  const int ml = lane & 15;
  const int kh = lane >> 4;

  const unsigned short* aBase = hb  + (size_t)(t0 + ar) * I_DIM + ah;
  const unsigned short* bBase = wtd + (size_t)(n0 + br) * I_DIM + bq;

  v8f acc[2][2] = {};

  auto stage = [&](int buf, int c) {
    const int e  = c >> 5;              // I_DIM/TK == 32
    const int k0 = (c & 31) * TK;
    const unsigned short* a = aBase + (size_t)e * T_TOK * I_DIM + k0;
    const unsigned short* b = bBase + (size_t)e * I_DIM * H_DIM + k0;
    async_b128(&As[buf][ar][ah],     a);
    async_b128(&As[buf][ar][ah + 8], a + 8);
    async_b128(&Bs[buf][br][bq],     b);
  };
  auto compute = [&](int buf) {
    v16bf aF[2], bF[2];
#pragma unroll
    for (int ms = 0; ms < 2; ++ms) {
      const unsigned short* p = &As[buf][wm * 32 + ms * 16 + ml][kh * 16];
      aF[ms] = make_frag(p, p + 8);
    }
#pragma unroll
    for (int ns = 0; ns < 2; ++ns) {
      const unsigned short* p = &Bs[buf][wn * 32 + ns * 16 + ml][kh * 16];
      bF[ns] = make_frag(p, p + 8);
    }
#pragma unroll
    for (int ms = 0; ms < 2; ++ms)
#pragma unroll
      for (int ns = 0; ns < 2; ++ns)
        acc[ms][ns] = __builtin_amdgcn_wmma_f32_16x16x32_bf16(
            false, aF[ms], false, bF[ns], (short)0, acc[ms][ns], false, false);
  };

  const int NC = NEXP * (I_DIM / TK);   // 320 chunks
  stage(0, 0);
  stage(1, 1);
  stage(2, 2);
  int c = 0;
  for (; c + 3 < NC; ++c) {             // steady state: 3 stages in flight
    wait_async_le<6>();                 // retire oldest stage (chunk c)
    __syncthreads();
    stage((c + 3) & 3, c + 3);          // prefetch 3 ahead
    compute(c & 3);
  }
  for (; c < NC; ++c) {                 // tail: full drain
    wait_async_le<0>();
    __syncthreads();
    compute(c & 3);
  }

  const int nl = lane & 15;
  const int mb = (lane >> 4) * 8;
#pragma unroll
  for (int ms = 0; ms < 2; ++ms)
#pragma unroll
    for (int ns = 0; ns < 2; ++ns) {
      int col = n0 + wn * 32 + ns * 16 + nl;
#pragma unroll
      for (int r = 0; r < 8; ++r) {
        int t = t0 + wm * 32 + ms * 16 + mb + r;
        out[(size_t)t * H_DIM + col] = acc[ms][ns][r];
      }
    }
}

// ================= host launcher =================
extern "C" void kernel_launch(void* const* d_in, const int* in_sizes, int n_in,
                              void* d_out, int out_size, void* d_ws, size_t ws_size,
                              hipStream_t stream) {
  (void)in_sizes; (void)n_in; (void)out_size; (void)ws_size;
  const float* x  = (const float*)d_in[0];
  const float* sg = (const float*)d_in[1];
  const float* su = (const float*)d_in[2];
  const float* sd = (const float*)d_in[3];
  const float* rg = (const float*)d_in[4];
  const float* ru = (const float*)d_in[5];
  const float* rd = (const float*)d_in[6];
  const float* rw = (const float*)d_in[7];
  const float* rb = (const float*)d_in[8];

  float* out   = (float*)d_out;
  float* usage = out + (size_t)T_TOK * H_DIM;

  char* ws = (char*)d_ws;
  size_t off = 0;
  const size_t AL = 255;
  unsigned short* xb = (unsigned short*)(ws + off);                // [T,H] bf16
  off += (size_t)T_TOK * H_DIM * 2;            off = (off + AL) & ~AL;
  float* comb = (float*)(ws + off);                                // [T,8]
  off += (size_t)T_TOK * 8 * 4;                off = (off + AL) & ~AL;
  unsigned short* hb = (unsigned short*)(ws + off);                // [NEXP,T,I]
  off += (size_t)NEXP * T_TOK * I_DIM * 2;     off = (off + AL) & ~AL;
  unsigned short* wtg = (unsigned short*)(ws + off);               // [NEXP,I,H]
  off += (size_t)NEXP * H_DIM * I_DIM * 2;     off = (off + AL) & ~AL;
  unsigned short* wtu = (unsigned short*)(ws + off);               // [NEXP,I,H]
  off += (size_t)NEXP * H_DIM * I_DIM * 2;     off = (off + AL) & ~AL;
  unsigned short* wtd = (unsigned short*)(ws + off);               // [NEXP,H,I]

  hipMemsetAsync(usage, 0, NROUTED * sizeof(float), stream);

  moe_router_k<<<T_TOK / 8, 256, 0, stream>>>(x, rw, rb, comb, usage);
  conv_bf16_k<<<(T_TOK * H_DIM / 4 + 255) / 256, 256, 0, stream>>>(
      x, xb, T_TOK * H_DIM / 4);
  wtrans_k<<<dim3(I_DIM / 32, H_DIM / 32, NEXP), 256, 0, stream>>>(
      sg, rg, wtg, H_DIM, I_DIM);
  wtrans_k<<<dim3(I_DIM / 32, H_DIM / 32, NEXP), 256, 0, stream>>>(
      su, ru, wtu, H_DIM, I_DIM);
  wtrans_k<<<dim3(H_DIM / 32, I_DIM / 32, NEXP), 256, 0, stream>>>(
      sd, rd, wtd, I_DIM, H_DIM);
  moe_gemm1_k<<<dim3(I_DIM / TN, T_TOK / TM, NEXP), 256, 0, stream>>>(
      xb, wtg, wtu, comb, hb);
  moe_gemm2_k<<<dim3(H_DIM / TN, T_TOK / TM), 256, 0, stream>>>(hb, wtd, out);
}
